// WindowAttention3D_21981642621142
// MI455X (gfx1250) — matrix operand order repaired and checked by reading, not compiled, not run on hardware
//
#include <hip/hip_runtime.h>
#include <hip/hip_bf16.h>

typedef __attribute__((ext_vector_type(16))) _Float16 v16h;
typedef __attribute__((ext_vector_type(8)))  _Float16 v8h;
typedef __attribute__((ext_vector_type(4)))  _Float16 v4h;
typedef __attribute__((ext_vector_type(8)))  float    v8f;

#define BWIN 256
#define NTOK 343
#define DIM  384
#define NH   12
#define HD   32
#define NPAD 352
#define MTOT (BWIN*NTOK)      /* 87808 = 16*5488, exact 16-row tiling */
#define QKVN (3*DIM)          /* 1152 */
#define QK_SCALE 0.17677669529663687f /* 1/sqrt(32) */

// ---- fragment loaders (layouts per CDNA5 ISA §7.12.2) ----

// A 16x32 f16 fragment. p = row base + 8*lo; K runs at +0..7 and +16..23.
__device__ inline v16h load_a_from_f16(const _Float16* __restrict__ p) {
  v8h a = *(const v8h*)p;
  v8h b = *(const v8h*)(p + 16);
  v16h A;
#pragma unroll
  for (int j = 0; j < 8; ++j) { A[j] = a[j]; A[j + 8] = b[j]; }
  return A;
}

// B 32x16 f16 fragment, column-per-lane. p = col base + 8*lo; K runs at +0..7 and +16..23.
__device__ inline v16h load_b_from_f16(const _Float16* __restrict__ p) {
  v8h a = *(const v8h*)p;
  v8h b = *(const v8h*)(p + 16);
  v16h B;
#pragma unroll
  for (int j = 0; j < 8; ++j) { B[j] = a[j]; B[j + 8] = b[j]; }
  return B;
}

// ---------------- Prep kernel A: x f32 -> f16 (same layout) ----------------
__global__ __launch_bounds__(256) void cvt_f32_to_f16_kernel(
    const float* __restrict__ src, _Float16* __restrict__ dst, long n4) {
  long i = (long)blockIdx.x * 256 + threadIdx.x;
  if (i < n4) {
    float4 v = ((const float4*)src)[i];
    v4h o;
    o[0] = (_Float16)v.x; o[1] = (_Float16)v.y;
    o[2] = (_Float16)v.z; o[3] = (_Float16)v.w;
    ((v4h*)dst)[i] = o;
  }
}

// ---------------- Prep kernel B: weight transpose f32 -> f16, wt[c][r] = w[r][c] ---
__global__ __launch_bounds__(256) void transpose_f32_to_f16_kernel(
    const float* __restrict__ w, _Float16* __restrict__ wt, int rows, int cols) {
  long idx = (long)blockIdx.x * 256 + threadIdx.x;
  if (idx < (long)rows * cols) {
    int r = (int)(idx / cols);
    int c = (int)(idx - (long)r * cols);
    wt[(size_t)c * rows + r] = (_Float16)w[idx];
  }
}

// ---------------- Kernel 1: QKV GEMM (16x32 tile/wave) + head-major scatter -----
__device__ inline void qkv_store(_Float16* __restrict__ q, _Float16* __restrict__ k,
                                 _Float16* __restrict__ vT,
                                 int col, int b, int tok, float val) {
  int sec = col / DIM;              // 0=q, 1=k, 2=v (tile never straddles: 32|384)
  int c = col - sec * DIM;
  int h = c >> 5, d = c & 31;
  size_t bh = (size_t)b * NH + h;
  if (sec == 0)      q [(bh * NPAD + tok) * HD + d] = (_Float16)(val * QK_SCALE);
  else if (sec == 1) k [(bh * NPAD + tok) * HD + d] = (_Float16)val;
  else               vT[(bh * HD + d) * NPAD + tok] = (_Float16)val;
}

__global__ __launch_bounds__(256) void qkv_gemm_kernel(
    const _Float16* __restrict__ xh, const _Float16* __restrict__ wT,
    const float* __restrict__ qkv_b,
    _Float16* __restrict__ q, _Float16* __restrict__ k, _Float16* __restrict__ vT)
{
  const int lane = threadIdx.x & 31;
  const int wave = threadIdx.x >> 5;
  const int lo = lane >> 4;
  const int ln = lane & 15;
  const int row0 = (blockIdx.y * 8 + wave) * 16;
  const int n0 = blockIdx.x * 32;

  const _Float16* abase  = xh + (size_t)(row0 + ln) * DIM + 8 * lo;
  const _Float16* b0base = wT + (size_t)(n0 + ln)      * DIM + 8 * lo;
  const _Float16* b1base = wT + (size_t)(n0 + 16 + ln) * DIM + 8 * lo;

  v8f acc0 = {}, acc1 = {};
  for (int kk = 0; kk < DIM / 32; ++kk) {
    const int k0 = kk * 32;
    if (kk + 1 < DIM / 32) {
      __builtin_prefetch(abase  + k0 + 32, 0, 3);
      __builtin_prefetch(b0base + k0 + 32, 0, 3);
      __builtin_prefetch(b1base + k0 + 32, 0, 3);
    }
    v16h A  = load_a_from_f16(abase  + k0);
    v16h B0 = load_b_from_f16(b0base + k0);
    v16h B1 = load_b_from_f16(b1base + k0);
    acc0 = __builtin_amdgcn_wmma_f32_16x16x32_f16(false, A, false, B0, (short)0, acc0, false, false);
    acc1 = __builtin_amdgcn_wmma_f32_16x16x32_f16(false, A, false, B1, (short)0, acc1, false, false);
  }

  const int col0 = n0 + ln;
  const int col1 = col0 + 16;
  const float bv0 = qkv_b[col0];
  const float bv1 = qkv_b[col1];
#pragma unroll
  for (int r = 0; r < 8; ++r) {
    int row = row0 + 8 * lo + r;               // always < MTOT (exact tiling)
    int b = row / NTOK;
    int tok = row - b * NTOK;
    qkv_store(q, k, vT, col0, b, tok, acc0[r] + bv0);
    qkv_store(q, k, vT, col1, b, tok, acc1[r] + bv1);
  }
}

// ---------------- Kernel 2: attention (S = QK^T, softmax+bias, O = PV) ----------
__global__ __launch_bounds__(32) void attn_kernel(
    const _Float16* __restrict__ q, const _Float16* __restrict__ kmat,
    const _Float16* __restrict__ vT,
    const float* __restrict__ bias_table, const int* __restrict__ rel_idx,
    _Float16* __restrict__ oh)
{
  __shared__ __align__(16) float    S[16][NPAD];
  __shared__ __align__(16) _Float16 P[16][NPAD];
  __shared__ float red[2][16];

  const int lane = threadIdx.x;
  const int lo = lane >> 4;
  const int ln = lane & 15;
  const int it = blockIdx.x;       // 0..21
  const int h  = blockIdx.y;       // 0..11
  const int b  = blockIdx.z;       // 0..255
  const int i0 = it * 16;
  const size_t bh = (size_t)b * NH + h;
  const _Float16* qb = q    + bh * (size_t)NPAD * HD;
  const _Float16* kb = kmat + bh * (size_t)NPAD * HD;
  const _Float16* vb = vT   + bh * (size_t)HD * NPAD;

  // Q fragment (16 rows x K=32 = full head dim), invariant over j tiles.
  v16h Aq = load_a_from_f16(qb + (size_t)(i0 + ln) * HD + 8 * lo);

  // Phase 1: S tiles via WMMA into LDS.
  for (int jt = 0; jt < NPAD / 16; ++jt) {
    int j0 = jt * 16;
    v16h B = load_b_from_f16(kb + (size_t)(j0 + ln) * HD + 8 * lo);
    v8f c = {};
    c = __builtin_amdgcn_wmma_f32_16x16x32_f16(false, Aq, false, B, (short)0, c, false, false);
#pragma unroll
    for (int r = 0; r < 8; ++r) S[8 * lo + r][j0 + ln] = c[r];
  }
  // Warm V^T tile while we do the softmax (lane i prefetches row i start).
  __builtin_prefetch(vb + (size_t)lane * NPAD, 0, 3);
  __syncthreads();

  // Phase 2: rel-pos bias + masked softmax. Thread = (row=ln, half=lo) over 176 cols.
  {
    const int row = ln;
    const int half = lo;
    const int i = i0 + row;
    const int jbeg = half * (NPAD / 2);
    const int irow = (i < NTOK) ? i : (NTOK - 1);
    const int* rp = rel_idx + (size_t)irow * NTOK;

    float mx = -3.0e38f;
    for (int jj = 0; jj < NPAD / 2; ++jj) {
      int j = jbeg + jj;
      float s;
      if (i < NTOK && j < NTOK)
        s = S[row][j] + bias_table[(size_t)rp[j] * NH + h];
      else
        s = -1.0e30f;
      S[row][j] = s;
      mx = fmaxf(mx, s);
    }
    red[half][row] = mx;
    __syncthreads();
    mx = fmaxf(red[0][row], red[1][row]);
    __syncthreads();

    float sum = 0.f;
    for (int jj = 0; jj < NPAD / 2; ++jj) {
      int j = jbeg + jj;
      float e = __expf(S[row][j] - mx);
      S[row][j] = e;
      sum += e;
    }
    red[half][row] = sum;
    __syncthreads();
    float inv = 1.0f / (red[0][row] + red[1][row]);
    for (int jj = 0; jj < NPAD / 2; ++jj) {
      int j = jbeg + jj;
      P[row][j] = (_Float16)(S[row][j] * inv);
    }
  }
  __syncthreads();

  // Phase 3: O = P @ V  (K loop over j, two n-tiles covering d = 0..31)
  v8f o0 = {}, o1 = {};
  for (int kt = 0; kt < NPAD / 32; ++kt) {
    int j0 = kt * 32;
    v16h Ap;
    {
      const _Float16* pp = &P[ln][j0 + 8 * lo];
      v8h a = *(const v8h*)pp;
      v8h bb = *(const v8h*)(pp + 16);
#pragma unroll
      for (int j = 0; j < 8; ++j) { Ap[j] = a[j]; Ap[j + 8] = bb[j]; }
    }
    v16h B0 = load_b_from_f16(vb + (size_t)ln        * NPAD + j0 + 8 * lo);
    v16h B1 = load_b_from_f16(vb + (size_t)(16 + ln) * NPAD + j0 + 8 * lo);
    o0 = __builtin_amdgcn_wmma_f32_16x16x32_f16(false, Ap, false, B0, (short)0, o0, false, false);
    o1 = __builtin_amdgcn_wmma_f32_16x16x32_f16(false, Ap, false, B1, (short)0, o1, false, false);
  }

#pragma unroll
  for (int r = 0; r < 8; ++r) {
    int i = i0 + 8 * lo + r;
    if (i < NTOK) {
      size_t o = ((size_t)b * NTOK + i) * DIM + (size_t)h * HD;
      oh[o + ln]      = (_Float16)o0[r];
      oh[o + 16 + ln] = (_Float16)o1[r];
    }
  }
}

// ---------------- Kernel 3: output projection (16x32 tile/wave) ----------------
__global__ __launch_bounds__(256) void proj_gemm_kernel(
    const _Float16* __restrict__ oh, const _Float16* __restrict__ wT,
    const float* __restrict__ proj_b, float* __restrict__ out)
{
  const int lane = threadIdx.x & 31;
  const int wave = threadIdx.x >> 5;
  const int lo = lane >> 4;
  const int ln = lane & 15;
  const int row0 = (blockIdx.y * 8 + wave) * 16;
  const int n0 = blockIdx.x * 32;

  const _Float16* abase  = oh + (size_t)(row0 + ln) * DIM + 8 * lo;
  const _Float16* b0base = wT + (size_t)(n0 + ln)      * DIM + 8 * lo;
  const _Float16* b1base = wT + (size_t)(n0 + 16 + ln) * DIM + 8 * lo;

  v8f acc0 = {}, acc1 = {};
  for (int kk = 0; kk < DIM / 32; ++kk) {
    const int k0 = kk * 32;
    if (kk + 1 < DIM / 32) {
      __builtin_prefetch(abase  + k0 + 32, 0, 3);
      __builtin_prefetch(b0base + k0 + 32, 0, 3);
      __builtin_prefetch(b1base + k0 + 32, 0, 3);
    }
    v16h A  = load_a_from_f16(abase  + k0);
    v16h B0 = load_b_from_f16(b0base + k0);
    v16h B1 = load_b_from_f16(b1base + k0);
    acc0 = __builtin_amdgcn_wmma_f32_16x16x32_f16(false, A, false, B0, (short)0, acc0, false, false);
    acc1 = __builtin_amdgcn_wmma_f32_16x16x32_f16(false, A, false, B1, (short)0, acc1, false, false);
  }

  const int col0 = n0 + ln;
  const int col1 = col0 + 16;
  const float bv0 = proj_b[col0];
  const float bv1 = proj_b[col1];
#pragma unroll
  for (int r = 0; r < 8; ++r) {
    int row = row0 + 8 * lo + r;               // always < MTOT
    out[(size_t)row * DIM + col0] = acc0[r] + bv0;
    out[(size_t)row * DIM + col1] = acc1[r] + bv1;
  }
}

extern "C" void kernel_launch(void* const* d_in, const int* in_sizes, int n_in,
                              void* d_out, int out_size, void* d_ws, size_t ws_size,
                              hipStream_t stream) {
  (void)in_sizes; (void)n_in; (void)out_size; (void)ws_size;
  const float* x          = (const float*)d_in[0];
  const float* qkv_w      = (const float*)d_in[1];
  const float* qkv_b      = (const float*)d_in[2];
  const float* proj_w     = (const float*)d_in[3];
  const float* proj_b     = (const float*)d_in[4];
  const float* bias_table = (const float*)d_in[5];
  const int*   rel_idx    = (const int*)d_in[6];
  float* out = (float*)d_out;

  // Workspace partition (all f16 elements)
  const size_t qn   = (size_t)BWIN * NH * NPAD * HD;  // q / k / vT each
  const size_t ohn  = (size_t)MTOT * DIM;             // attention output (pre-proj)
  const size_t xn   = (size_t)MTOT * DIM;             // f16 copy of x
  _Float16* q   = (_Float16*)d_ws;
  _Float16* k   = q  + qn;
  _Float16* vT  = k  + qn;
  _Float16* oh  = vT + qn;
  _Float16* xh  = oh + ohn;
  _Float16* qwT = xh + xn;                            // 1152 x 384
  _Float16* pwT = qwT + (size_t)QKVN * DIM;           // 384 x 384

  // Prep: x -> f16, weights -> transposed f16
  {
    long n4 = (long)MTOT * DIM / 4;                   // 8,429,568
    cvt_f32_to_f16_kernel<<<dim3((unsigned)((n4 + 255) / 256)), 256, 0, stream>>>(x, xh, n4);
    long wq = (long)DIM * QKVN;                       // 442,368
    transpose_f32_to_f16_kernel<<<dim3((unsigned)((wq + 255) / 256)), 256, 0, stream>>>(
        qkv_w, qwT, DIM, QKVN);
    long wp = (long)DIM * DIM;                        // 147,456
    transpose_f32_to_f16_kernel<<<dim3((unsigned)((wp + 255) / 256)), 256, 0, stream>>>(
        proj_w, pwT, DIM, DIM);
  }

  // QKV: N tiles = 1152/32 = 36; M groups = 5488/8 = 686 (8 waves/block)
  qkv_gemm_kernel<<<dim3(QKVN / 32, (MTOT / 16) / 8), 256, 0, stream>>>(
      xh, qwT, qkv_b, q, k, vT);

  // Attention: (22 i-tiles) x (12 heads) x (256 windows), one wave each
  attn_kernel<<<dim3(NPAD / 16, NH, BWIN), 32, 0, stream>>>(
      q, k, vT, bias_table, rel_idx, oh);

  // Projection: N tiles = 384/32 = 12
  proj_gemm_kernel<<<dim3(DIM / 32, (MTOT / 16) / 8), 256, 0, stream>>>(
      oh, pwT, proj_b, out);
}
